// CovKernelCometric_55508157333789
// MI455X (gfx1250) — matrix operand order repaired and checked by reading, not compiled, not run on hardware
//
#include <hip/hip_runtime.h>
#include <hip/hip_bf16.h>
#include <math.h>

// CDNA5 wave32 WMMA fragment types
typedef __attribute__((ext_vector_type(16))) __bf16 v16bf;
typedef __attribute__((ext_vector_type(8)))  float  v8f;

#define A_PARAM  1.0f
#define REG_COEF 1e-3f
#define B_N 2048
#define K_N 1024
#define D_N 64

__device__ inline v8f wmma_bf16(v16bf a, v16bf b, v8f c) {
    // D = A(16x32 bf16) x B(32x16 bf16) + C(16x16 f32)
    return __builtin_amdgcn_wmma_f32_16x16x32_bf16(
        false, a, false, b, (short)0, c, false, false);
}

// ---------------------------------------------------------------------------
// rho: per-keypoint min distance to nearest other keypoint
// ---------------------------------------------------------------------------
__global__ __launch_bounds__(256) void k_rho_min(const float* __restrict__ c,
                                                 float* __restrict__ rmin) {
    __shared__ float ci[64];
    __shared__ float red[256];
    int i = blockIdx.x;
    int t = threadIdx.x;
    if (t < 64) ci[t] = c[(long)i * 64 + t];
    __syncthreads();
    float best = INFINITY;
    for (int j = t; j < K_N; j += 256) {
        if (j == i) continue;
        float s = 0.f;
        #pragma unroll 8
        for (int dd = 0; dd < 64; ++dd) {
            float df = ci[dd] - c[(long)j * 64 + dd];
            s += df * df;
        }
        best = fminf(best, s);
    }
    red[t] = best;
    __syncthreads();
    for (int off = 128; off > 0; off >>= 1) {
        if (t < off) red[t] = fminf(red[t], red[t + off]);
        __syncthreads();
    }
    if (t == 0) rmin[i] = sqrtf(fmaxf(red[0], 0.f));
}

// deterministic tree-sum of the 1024 mins -> inv2rho2 scalar
__global__ __launch_bounds__(256) void k_rho_reduce(const float* __restrict__ rmin,
                                                    float* __restrict__ scal) {
    __shared__ float red[256];
    int t = threadIdx.x;
    float s = 0.f;
    for (int i = t; i < K_N; i += 256) s += rmin[i];
    red[t] = s;
    __syncthreads();
    for (int off = 128; off > 0; off >>= 1) {
        if (t < off) red[t] += red[t + off];
        __syncthreads();
    }
    if (t == 0) {
        float rho = A_PARAM * (red[0] / (float)K_N);
        scal[0] = 1.0f / (2.0f * rho * rho);
    }
}

// ---------------------------------------------------------------------------
// Pack G into per-lane B-fragment layout (bf16 hi/lo split).
// Pass-1 layout: [k][chunk(2)][ntile(4)][lane(32)][16 bf16]
//   fragment element i  <->  G_k[ d = (lane<16?0:8)+32*chunk+i+(i<8?0:8) , e = ntile*16+lane%16 ]
// ---------------------------------------------------------------------------
__global__ __launch_bounds__(256) void k_pack1(const float* __restrict__ g,
                                               __bf16* __restrict__ ph,
                                               __bf16* __restrict__ pl) {
    int gid   = blockIdx.x * 256 + threadIdx.x;   // k*256 + chunk*128 + nt*32 + lane
    int lane  = gid & 31;
    int nt    = (gid >> 5) & 3;
    int chunk = (gid >> 7) & 1;
    int k     = gid >> 8;
    int e     = nt * 16 + (lane & 15);
    int dbase = ((lane < 16) ? 0 : 8) + chunk * 32;
    long base = (long)gid * 16;
    #pragma unroll
    for (int i = 0; i < 16; ++i) {
        float x = g[(long)k * 4096 + (long)(dbase + i + (i < 8 ? 0 : 8)) * 64 + e];
        __bf16 hi = (__bf16)x;
        ph[base + i] = hi;
        pl[base + i] = (__bf16)(x - (float)hi);
    }
}

// Pass-2 layout: [kc(32)][ntile(256)][lane(32)][16 bf16]
//   fragment element i <-> Gv[ kp = kc*32+(lane<16?0:8)+i+(i<8?0:8) , n = ntile*16+lane%16 ]
__global__ __launch_bounds__(256) void k_pack2(const float* __restrict__ g,
                                               __bf16* __restrict__ ph,
                                               __bf16* __restrict__ pl) {
    int gid  = blockIdx.x * 256 + threadIdx.x;    // kc*8192 + nt*32 + lane
    int lane = gid & 31;
    int nt   = (gid >> 5) & 255;
    int kc   = gid >> 13;
    int n     = nt * 16 + (lane & 15);
    int kbase = kc * 32 + ((lane < 16) ? 0 : 8);
    long base = (long)gid * 16;
    #pragma unroll
    for (int i = 0; i < 16; ++i) {
        float x = g[(long)(kbase + i + (i < 8 ? 0 : 8)) * 4096 + n];
        __bf16 hi = (__bf16)x;
        ph[base + i] = hi;
        pl[base + i] = (__bf16)(x - (float)hi);
    }
}

// ---------------------------------------------------------------------------
// Pass 1: quad[b,k] = (q_b - c_k)^T G_k (q_b - c_k);  W = exp(-quad*inv2rho2)
// Block: 8 waves x 16 b-rows = 128 b-rows; 64 k per block.
// Packed G_k fragments (16 KB hi+lo) are staged into LDS once per block per k
// (8x reduction of global/L2 traffic vs per-wave loads), with a global_prefetch
// of the next k's fragment stream issued during the compute phase.
// Per (wave,k): T(16x64) = Y @ G_k via 24 WMMAs (hi/lo split), quad = rowsum(T.*Y).
// ---------------------------------------------------------------------------
__global__ __launch_bounds__(256) void k_quad(const float* __restrict__ q,
                                              const float* __restrict__ c,
                                              const v16bf* __restrict__ p1h,
                                              const v16bf* __restrict__ p1l,
                                              const float* __restrict__ scal,
                                              float* __restrict__ W) {
    __shared__ float qs[128 * 68];   // row stride 68 -> conflict-free
    __shared__ v16bf sgh[256];       // staged G_k hi fragments (8 KB)
    __shared__ v16bf sgl[256];       // staged G_k lo fragments (8 KB)
    int t = threadIdx.x;
    int wave = t >> 5, lane = t & 31;
    int b0 = blockIdx.x * 128;
    int k0 = blockIdx.y * 64;

    for (int idx = t; idx < 128 * 64; idx += 256) {
        int r = idx >> 6, e = idx & 63;
        qs[r * 68 + e] = q[(long)(b0 + r) * 64 + e];
    }

    float inv2rho2 = scal[0];
    int   mrow  = lane & 15;               // A-frag row (local)
    int   base0 = (lane < 16) ? 0 : 8;     // A-frag & C-frag half offset
    const float* qrow = &qs[(wave * 16 + mrow) * 68];

    for (int kk = 0; kk < 64; ++kk) {
        int k = k0 + kk;
        const float* ck = &c[(long)k * 64];

        // stage packed G_k fragments into LDS (one v16bf per thread per buffer)
        __syncthreads();   // previous iteration's fragment reads done
        sgh[t] = p1h[(long)k * 256 + t];
        sgl[t] = p1l[(long)k * 256 + t];
        if (kk + 1 < 64) {  // prefetch next k's fragment stream into caches
            __builtin_prefetch(&p1h[(long)(k + 1) * 256 + t], 0, 3);
            __builtin_prefetch(&p1l[(long)(k + 1) * 256 + t], 0, 3);
        }
        __syncthreads();

        // Build A fragments: Y = q_row - c_k, split into bf16 hi/lo
        v16bf ahi[2], alo[2];
        #pragma unroll
        for (int chunk = 0; chunk < 2; ++chunk) {
            int e0 = chunk * 32 + base0;
            #pragma unroll
            for (int i = 0; i < 8; ++i) {
                float y0 = qrow[e0 + i]      - ck[e0 + i];
                float y1 = qrow[e0 + 16 + i] - ck[e0 + 16 + i];
                __bf16 h0 = (__bf16)y0;
                __bf16 h1 = (__bf16)y1;
                ahi[chunk][i]     = h0;  alo[chunk][i]     = (__bf16)(y0 - (float)h0);
                ahi[chunk][8 + i] = h1;  alo[chunk][8 + i] = (__bf16)(y1 - (float)h1);
            }
        }

        v8f acc[4];
        #pragma unroll
        for (int nt = 0; nt < 4; ++nt) acc[nt] = (v8f){0.f,0.f,0.f,0.f,0.f,0.f,0.f,0.f};

        #pragma unroll
        for (int nt = 0; nt < 4; ++nt) {
            #pragma unroll
            for (int chunk = 0; chunk < 2; ++chunk) {
                int fi = (chunk * 4 + nt) * 32 + lane;
                v16bf bh = sgh[fi];
                v16bf bl = sgl[fi];
                acc[nt] = wmma_bf16(ahi[chunk], bh, acc[nt]);
                acc[nt] = wmma_bf16(ahi[chunk], bl, acc[nt]);
                acc[nt] = wmma_bf16(alo[chunk], bh, acc[nt]);
            }
        }

        // quad rowsum: C layout -> (M = r+base0, e = nt*16 + lane%16)
        float pacc[8];
        #pragma unroll
        for (int r = 0; r < 8; ++r) pacc[r] = 0.f;
        #pragma unroll
        for (int nt = 0; nt < 4; ++nt) {
            int e = nt * 16 + (lane & 15);
            float cv = ck[e];
            #pragma unroll
            for (int r = 0; r < 8; ++r) {
                int m = r + base0;
                pacc[r] += acc[nt][r] * (qs[(wave * 16 + m) * 68 + e] - cv);
            }
        }
        // reduce over e within each 16-lane half (xor 1..8 never crosses halves)
        #pragma unroll
        for (int off = 8; off >= 1; off >>= 1) {
            #pragma unroll
            for (int r = 0; r < 8; ++r) pacc[r] += __shfl_xor(pacc[r], off, 32);
        }
        if ((lane & 15) == 0) {
            int rowbase = b0 + wave * 16 + base0;
            #pragma unroll
            for (int r = 0; r < 8; ++r) {
                float wv = __expf(-pacc[r] * inv2rho2);
                W[(long)(rowbase + r) * K_N + k] = wv;
            }
        }
    }
}

// ---------------------------------------------------------------------------
// Pass 2: out(2048 x 4096) = W(2048 x 1024) @ Gv(1024 x 4096) + REG_COEF * I
// Block: 8 waves; wave -> one 16x16 out tile, inner K in 32-wide chunks.
// ---------------------------------------------------------------------------
__global__ __launch_bounds__(256) void k_out(const float* __restrict__ W,
                                             const v16bf* __restrict__ p2h,
                                             const v16bf* __restrict__ p2l,
                                             float* __restrict__ out) {
    int t = threadIdx.x, wave = t >> 5, lane = t & 31;
    int b0    = blockIdx.y * 16;
    int ntile = blockIdx.x * 8 + wave;
    int mrow  = lane & 15;
    int base0 = (lane < 16) ? 0 : 8;
    const float* wrow = &W[(long)(b0 + mrow) * K_N];

    v8f acc = (v8f){0.f,0.f,0.f,0.f,0.f,0.f,0.f,0.f};
    for (int kc = 0; kc < 32; ++kc) {
        v16bf ahi, alo;
        int kb = kc * 32 + base0;
        #pragma unroll
        for (int i = 0; i < 8; ++i) {
            float x0 = wrow[kb + i];
            float x1 = wrow[kb + 16 + i];
            __bf16 h0 = (__bf16)x0, h1 = (__bf16)x1;
            ahi[i]     = h0;  alo[i]     = (__bf16)(x0 - (float)h0);
            ahi[8 + i] = h1;  alo[8 + i] = (__bf16)(x1 - (float)h1);
        }
        long bi = ((long)kc * 256 + ntile) * 32 + lane;
        v16bf bh = p2h[bi];
        v16bf bl = p2l[bi];
        acc = wmma_bf16(ahi, bh, acc);
        acc = wmma_bf16(ahi, bl, acc);
        acc = wmma_bf16(alo, bh, acc);
    }

    int nloc = ntile * 16 + (lane & 15);
    bool diag = (nloc % 65) == 0;        // n = 65*j are the d x d diagonal slots
    #pragma unroll
    for (int r = 0; r < 8; ++r) {
        int m = b0 + base0 + r;
        float v = acc[r];
        if (diag) v += REG_COEF;
        out[(long)m * 4096 + nloc] = v;
    }
}

// ---------------------------------------------------------------------------
extern "C" void kernel_launch(void* const* d_in, const int* in_sizes, int n_in,
                              void* d_out, int out_size, void* d_ws, size_t ws_size,
                              hipStream_t stream) {
    (void)in_sizes; (void)n_in; (void)out_size; (void)ws_size;
    const float* q = (const float*)d_in[0];   // (2048, 64)
    const float* c = (const float*)d_in[1];   // (1024, 64)
    const float* g = (const float*)d_in[2];   // (1024, 64, 64)
    float* out = (float*)d_out;               // (2048, 64, 64)

    char* ws = (char*)d_ws;
    float*  scal = (float*)(ws + 0);                       // inv2rho2
    float*  rmin = (float*)(ws + 256);                     // 1024 f32
    float*  W    = (float*)(ws + (1ull << 20));            // 8 MB
    __bf16* p1h  = (__bf16*)(ws + (16ull << 20));          // 8 MB
    __bf16* p1l  = (__bf16*)(ws + (24ull << 20));          // 8 MB
    __bf16* p2h  = (__bf16*)(ws + (32ull << 20));          // 8 MB
    __bf16* p2l  = (__bf16*)(ws + (40ull << 20));          // 8 MB

    k_rho_min   <<<K_N, 256, 0, stream>>>(c, rmin);
    k_rho_reduce<<<1,   256, 0, stream>>>(rmin, scal);
    k_pack1     <<<1024, 256, 0, stream>>>(g, p1h, p1l);
    k_pack2     <<<1024, 256, 0, stream>>>(g, p2h, p2l);
    k_quad      <<<dim3(B_N / 128, K_N / 64), 256, 0, stream>>>(
                    q, c, (const v16bf*)p1h, (const v16bf*)p1l, scal, W);
    k_out       <<<dim3(4096 / 128, B_N / 16), 256, 0, stream>>>(
                    W, (const v16bf*)p2h, (const v16bf*)p2l, out);
}
